// BarrierNet_30305289241224
// MI455X (gfx1250) — compile-verified, matrix operand order checked by reading, not run on hardware
//
#include <hip/hip_runtime.h>
#include <hip/hip_bf16.h>

typedef _Float16 v8h  __attribute__((ext_vector_type(8)));
typedef _Float16 v16h __attribute__((ext_vector_type(16)));
typedef float    v8f  __attribute__((ext_vector_type(8)));

#define B_ROWS   32768
#define H1       2048
#define H21      1024

// ---------------------------------------------------------------------------
// Kernel 1: h = relu(x @ W1 + b1), f32 compute, f16 store.
// Each block handles 8 batch rows so each W1 column load is reused 8x.
// ---------------------------------------------------------------------------
__global__ __launch_bounds__(256) void bn_h_kernel(
    const float* __restrict__ x, const float* __restrict__ W1,
    const float* __restrict__ b1, _Float16* __restrict__ H)
{
    __shared__ float xs[8][8];
    const int b0  = blockIdx.x * 8;
    const int tid = threadIdx.x;
    if (tid < 64) xs[tid >> 3][tid & 7] = x[(long)(b0 + (tid >> 3)) * 8 + (tid & 7)];
    __syncthreads();

    for (int j = tid; j < H1; j += 256) {
        float w[8];
#pragma unroll
        for (int k = 0; k < 8; ++k) w[k] = W1[k * H1 + j];
        const float bias = b1[j];
#pragma unroll
        for (int r = 0; r < 8; ++r) {
            float acc = bias;
#pragma unroll
            for (int k = 0; k < 8; ++k) acc = fmaf(xs[r][k], w[k], acc);
            acc = acc > 0.0f ? acc : 0.0f;
            H[(long)(b0 + r) * H1 + j] = (_Float16)acc;
        }
    }
}

// ---------------------------------------------------------------------------
// Kernel 2: W21t[n][k] = (f16)W21[k][n]  (LDS-tiled transpose + convert)
// so WMMA B fragments read contiguous K per lane.
// ---------------------------------------------------------------------------
__global__ __launch_bounds__(256) void bn_w21t_kernel(
    const float* __restrict__ W21, _Float16* __restrict__ W21t)
{
    __shared__ float tile[32][33];
    const int nt = blockIdx.x;            // 0..31  over N (1024/32)
    const int kt = blockIdx.y;            // 0..63  over K (2048/32)
    const int tx = threadIdx.x & 31;
    const int ty = threadIdx.x >> 5;      // 0..7
#pragma unroll
    for (int i = 0; i < 4; ++i) {
        const int k = kt * 32 + ty + i * 8;
        const int n = nt * 32 + tx;
        tile[ty + i * 8][tx] = W21[(long)k * H21 + n];
    }
    __syncthreads();
#pragma unroll
    for (int i = 0; i < 4; ++i) {
        const int n = nt * 32 + ty + i * 8;
        const int k = kt * 32 + tx;
        W21t[(long)n * H1 + k] = (_Float16)tile[tx][ty + i * 8];
    }
}

// ---------------------------------------------------------------------------
// 16-bit WMMA operand fragment loader (row-major [16 x K] source, stride H1):
// lanes 0-15 hold row (lane&15), K {0..7, 16..23}; lanes 16-31 K {8..15, 24..31}.
// Two 16-byte loads per lane.
// ---------------------------------------------------------------------------
__device__ __forceinline__ v16h load_frag16(const _Float16* __restrict__ base,
                                            int k0, int lane)
{
    const int r   = lane & 15;
    const int sel = (lane >> 4) & 1;
    const _Float16* p = base + (long)r * H1 + k0 + sel * 8;
    v8h lo = *(const v8h*)(p);
    v8h hi = *(const v8h*)(p + 16);
    v16h f;
#pragma unroll
    for (int i = 0; i < 8; ++i) { f[i] = lo[i]; f[i + 8] = hi[i]; }
    return f;
}

// ---------------------------------------------------------------------------
// Kernel 3: fused  a21 = relu(H @ W21 + b21);  x31 = a21 @ W31 + b31;
//           u = clip(-(x31*ostd+omean), lower, upper); out = (u-omean)/ostd
// Each wave: 32-row strip, processed as 2 M-tiles x 4 N-tiles register
// blocking -> 8 WMMAs per 6 fragment loads (1.5 b128 loads per WMMA).
// W31 contraction (N_CL=2) folded into the per-N-group epilogue, so only
// 32 accumulator VGPRs persist across the N loop.
// ---------------------------------------------------------------------------
__global__ __launch_bounds__(256) void bn_main_kernel(
    const _Float16* __restrict__ H, const _Float16* __restrict__ W21t,
    const float* __restrict__ b21, const float* __restrict__ W31,
    const float* __restrict__ b31,
    const float* __restrict__ omean, const float* __restrict__ ostd,
    const float* __restrict__ s0, const float* __restrict__ s1,
    const float* __restrict__ s2, const float* __restrict__ s3,
    float* __restrict__ out)
{
    const int lane = threadIdx.x & 31;
    const int wave = threadIdx.x >> 5;
    const int row0 = blockIdx.x * 256 + wave * 32;        // 32 rows per wave
    const _Float16* A0 = H + (long)row0 * H1;             // M-tile 0: rows 0..15
    const _Float16* A1 = A0 + (long)16 * H1;              // M-tile 1: rows 16..31

    // Per-row W31 partials: [M-tile][output channel][C-vgpr row]
    float acc[2][2][8];
#pragma unroll
    for (int mt = 0; mt < 2; ++mt)
#pragma unroll
        for (int ch = 0; ch < 2; ++ch)
#pragma unroll
            for (int r = 0; r < 8; ++r) acc[mt][ch][r] = 0.0f;

    const int n = lane & 15;

    for (int j = 0; j < H21 / 64; ++j) {                  // 16 groups of 4 N-tiles
        const _Float16* Bg = W21t + (long)(j * 64) * H1;

        // Prefetch first lines of the next N-group (global_prefetch_b8).
        if (j + 1 < H21 / 64) {
            const _Float16* pf = W21t + (long)((j + 1) * 64 + lane) * H1;
            __builtin_prefetch(pf, 0, 1);
        }

        v8f c[2][4];
#pragma unroll
        for (int mt = 0; mt < 2; ++mt)
#pragma unroll
            for (int nt = 0; nt < 4; ++nt) c[mt][nt] = (v8f){};

#pragma unroll 2
        for (int kk = 0; kk < H1 / 32; ++kk) {            // 64 K-steps
            const int k0 = kk * 32;
            v16h a[2], b[4];
            a[0] = load_frag16(A0, k0, lane);
            a[1] = load_frag16(A1, k0, lane);
#pragma unroll
            for (int nt = 0; nt < 4; ++nt)
                b[nt] = load_frag16(Bg + (long)(nt * 16) * H1, k0, lane);
#pragma unroll
            for (int mt = 0; mt < 2; ++mt)
#pragma unroll
                for (int nt = 0; nt < 4; ++nt)
                    c[mt][nt] = __builtin_amdgcn_wmma_f32_16x16x32_f16(
                        false, a[mt], false, b[nt], (short)0, c[mt][nt], false, false);
        }

        // C layout: VGPR r -> row M=r (lanes 0-15) / M=r+8 (lanes 16-31), N=lane&15
        float bi[4], w0[4], w1[4];
#pragma unroll
        for (int nt = 0; nt < 4; ++nt) {
            const int ng = j * 64 + nt * 16 + n;
            bi[nt] = b21[ng];
            w0[nt] = W31[2 * ng + 0];
            w1[nt] = W31[2 * ng + 1];
        }
#pragma unroll
        for (int mt = 0; mt < 2; ++mt)
#pragma unroll
            for (int r = 0; r < 8; ++r) {
                float s0v = acc[mt][0][r];
                float s1v = acc[mt][1][r];
#pragma unroll
                for (int nt = 0; nt < 4; ++nt) {
                    float v = c[mt][nt][r] + bi[nt];
                    v = v > 0.0f ? v : 0.0f;
                    s0v = fmaf(v, w0[nt], s0v);
                    s1v = fmaf(v, w1[nt], s1v);
                }
                acc[mt][0][r] = s0v;
                acc[mt][1][r] = s1v;
            }
    }

    // Reduce across the 16 N-lanes within each half-wave (rows differ per half).
#pragma unroll
    for (int mt = 0; mt < 2; ++mt)
#pragma unroll
        for (int ch = 0; ch < 2; ++ch)
#pragma unroll
            for (int r = 0; r < 8; ++r)
                for (int off = 8; off >= 1; off >>= 1)
                    acc[mt][ch][r] += __shfl_xor(acc[mt][ch][r], off, 16);

    if (n == 0) {
        const float om0 = omean[0], om1 = omean[1];
        const float os0 = ostd[0],  os1 = ostd[1];
        // upper = [OMEGA+s2, ACCEL+s0], lower = [-(OMEGA+s3), -(ACCEL+s1)], limits = 1.0
        const float up0 = 1.0f + s2[0], up1 = 1.0f + s0[0];
        const float lo0 = -(1.0f + s3[0]), lo1 = -(1.0f + s1[0]);
        const float bb0 = b31[0], bb1 = b31[1];
        const int half8 = (lane >> 4) ? 8 : 0;
#pragma unroll
        for (int mt = 0; mt < 2; ++mt) {
            const int mbase = row0 + mt * 16 + half8;
#pragma unroll
            for (int r = 0; r < 8; ++r) {
                const int m = mbase + r;
                const float a0 = fmaf(acc[mt][0][r] + bb0, os0, om0);
                const float a1 = fmaf(acc[mt][1][r] + bb1, os1, om1);
                const float u0 = fminf(fmaxf(-a0, lo0), up0);
                const float u1 = fminf(fmaxf(-a1, lo1), up1);
                out[m * 2 + 0] = (u0 - om0) / os0;
                out[m * 2 + 1] = (u1 - om1) / os1;
            }
        }
    }
}

// ---------------------------------------------------------------------------
extern "C" void kernel_launch(void* const* d_in, const int* in_sizes, int n_in,
                              void* d_out, int out_size, void* d_ws, size_t ws_size,
                              hipStream_t stream)
{
    (void)in_sizes; (void)n_in; (void)out_size; (void)ws_size;

    const float* x    = (const float*)d_in[0];
    const float* W1   = (const float*)d_in[1];
    const float* b1   = (const float*)d_in[2];
    const float* W21  = (const float*)d_in[3];
    const float* b21  = (const float*)d_in[4];
    const float* W31  = (const float*)d_in[5];
    const float* b31  = (const float*)d_in[6];
    // d_in[7..10] = W22/b22/W32/b32: dead branch (x32 never reaches the output)
    // d_in[11..12] = input_mean/std: dead (x0 unused)
    const float* omean = (const float*)d_in[13];
    const float* ostd  = (const float*)d_in[14];
    const float* s0 = (const float*)d_in[15];
    const float* s1 = (const float*)d_in[16];
    const float* s2 = (const float*)d_in[17];
    const float* s3 = (const float*)d_in[18];
    // d_in[19] = sgn: unused by the output

    float* out = (float*)d_out;

    _Float16* Hbuf  = (_Float16*)d_ws;                                    // 128 MiB
    _Float16* W21t  = (_Float16*)((char*)d_ws + (size_t)B_ROWS * H1 * 2); // + 4 MiB

    bn_h_kernel<<<B_ROWS / 8, 256, 0, stream>>>(x, W1, b1, Hbuf);
    bn_w21t_kernel<<<dim3(H21 / 32, H1 / 32), 256, 0, stream>>>(W21, W21t);
    bn_main_kernel<<<B_ROWS / 256, 256, 0, stream>>>(
        Hbuf, W21t, b21, W31, b31, omean, ostd, s0, s1, s2, s3, out);
}
